// AdaLoRAMLPWithBase_2199023256221
// MI455X (gfx1250) — matrix operand order repaired and checked by reading, not compiled, not run on hardware
//
#include <hip/hip_runtime.h>
#include <cstddef>

#define B_SZ   128
#define D_SZ   1024
#define ADA_SZ 1024
#define R_SZ   8
#define WCOLS  (4 * D_SZ * R_SZ)   // 32768

typedef __attribute__((ext_vector_type(16))) __bf16 v16bf;
typedef __attribute__((ext_vector_type(8)))  float  v8f;

__device__ __forceinline__ float gelu_exact(float v) {
  return 0.5f * v * (1.0f + erff(v * 0.70710678118654752f));
}

enum { EPI_BIAS = 0, EPI_BIAS_GELU = 1, EPI_LORA_GELU = 2, EPI_LORA_RESID = 3 };

// ---------------------------------------------------------------------------
// WMMA GEMM: C[128,N] = epilogue(A[128,K] * B[K,N])
// Block = 256 threads (8 wave32). Block tile = 128(M: all rows) x 16(N strip);
// wave w owns rows [16w, 16w+16). K-step 64 -> two v_wmma_f32_16x16x32_bf16
// per stage. B is staged TRANSPOSED in LDS (sBt[n][k]) and A row-major
// (sA[m][k]) so every lane's 16-element fragment is two contiguous 16B chunks
// -> ds_load_b128 pairs instead of scattered gathers.
// Each B element is read by exactly ONE block => the 128 MB W2 streams from
// HBM once (~5.5 us at 23.3 TB/s); A re-reads hit L2.
// BT=true reads B as row-major [N,K] (multiply by B^T).
// ---------------------------------------------------------------------------
template <int EPI, bool BT>
__global__ void __launch_bounds__(256) gemm_bf16_wmma(
    const float* __restrict__ A, int lda,
    const float* __restrict__ Bm, int ldb,
    const float* __restrict__ bias,
    float* __restrict__ C, int ldc,
    int K,
    const float* __restrict__ lw, int lws,      // LoRA factor block [128, N*8]
    const float* __restrict__ tvec,             // [128, 8]
    const float* __restrict__ resid)            // [128, ldc]
{
  __shared__ __align__(16) __bf16 sA[128][72];   // 128 x 64 (+pad, 16B-mult stride)
  __shared__ __align__(16) __bf16 sBt[16][72];   // transposed: [n][k], 16 x 64

  const int tid  = threadIdx.x;
  const int lane = tid & 31;
  const int wave = tid >> 5;
  const int n0   = blockIdx.x * 16;

  const int arow = tid >> 1;           // 0..127
  const int acol = (tid & 1) * 32;     // 0 or 32

  v8f acc = {};

  for (int k0 = 0; k0 < K; k0 += 64) {
    __syncthreads();
    // ---- stage A tile 128x64: fp32 -> bf16, b128 LDS stores ----
    {
      const float* asrc = &A[(size_t)arow * lda + k0 + acol];
      #pragma unroll
      for (int i = 0; i < 4; ++i) {
        const float4 v0 = *(const float4*)&asrc[i * 8];
        const float4 v1 = *(const float4*)&asrc[i * 8 + 4];
        __bf16 p[8] = {(__bf16)v0.x, (__bf16)v0.y, (__bf16)v0.z, (__bf16)v0.w,
                       (__bf16)v1.x, (__bf16)v1.y, (__bf16)v1.z, (__bf16)v1.w};
        *(uint4*)&sA[arow][acol + i * 8] = *(const uint4*)p;
      }
      if (k0 + 64 < K) __builtin_prefetch(&asrc[64], 0, 0);
    }
    // ---- stage B tile 64x16 transposed into sBt[n][k] ----
    if (!BT) {
      const int k  = tid >> 2;             // 0..63
      const int ng = (tid & 3) * 4;        // 0,4,8,12
      const float4 bv = *(const float4*)&Bm[(size_t)(k0 + k) * ldb + n0 + ng];
      sBt[ng + 0][k] = (__bf16)bv.x;
      sBt[ng + 1][k] = (__bf16)bv.y;
      sBt[ng + 2][k] = (__bf16)bv.z;
      sBt[ng + 3][k] = (__bf16)bv.w;
      if (k0 + 64 < K)  // global_prefetch_b8 on next streaming tile
        __builtin_prefetch(&Bm[(size_t)(k0 + 64 + k) * ldb + n0 + ng], 0, 0);
    } else {
      const int n  = tid >> 4;             // 0..15
      const int kg = (tid & 15) * 4;       // 0..60
      const float4 bv = *(const float4*)&Bm[(size_t)(n0 + n) * ldb + k0 + kg];
      __bf16 p[4] = {(__bf16)bv.x, (__bf16)bv.y, (__bf16)bv.z, (__bf16)bv.w};
      *(uint2*)&sBt[n][kg] = *(const uint2*)p;
      if (k0 + 64 < K)
        __builtin_prefetch(&Bm[(size_t)(n0 + n) * ldb + k0 + 64 + kg], 0, 0);
    }
    __syncthreads();

    // ---- fragments: contiguous b128 pairs per CDNA5 wave32 WMMA layouts ----
    // A 16x32: lane(m=lane&15, kh=lane>>4): K = {8kh..8kh+7} U {16+8kh..+7}
    // B 32x16: lane(n=lane&15, kh):         K = {8kh..8kh+7} U {16+8kh..+7}
    const int mr = wave * 16 + (lane & 15);
    const int nn = lane & 15;
    const int kh = lane >> 4;
    #pragma unroll
    for (int s = 0; s < 2; ++s) {
      union { v16bf v; uint4 q[2]; } fa, fb;
      fa.q[0] = *(const uint4*)&sA[mr][s * 32 + 8 * kh];
      fa.q[1] = *(const uint4*)&sA[mr][s * 32 + 16 + 8 * kh];
      fb.q[0] = *(const uint4*)&sBt[nn][s * 32 + 8 * kh];
      fb.q[1] = *(const uint4*)&sBt[nn][s * 32 + 16 + 8 * kh];
      acc = __builtin_amdgcn_wmma_f32_16x16x32_bf16(false, fa.v, false, fb.v,
                                                    (short)0, acc, false, false);
    }
  }

  // ---- epilogue: C/D element r lives at (M = 16*wave + r + 8*kh, N=lane&15)
  const int kh   = lane >> 4;
  const int ncol = n0 + (lane & 15);
  #pragma unroll
  for (int r = 0; r < 8; ++r) {
    const int mrow = wave * 16 + r + 8 * kh;
    float v = acc[r];
    if (EPI == EPI_BIAS || EPI == EPI_BIAS_GELU) v += bias[ncol];
    if (EPI == EPI_LORA_GELU || EPI == EPI_LORA_RESID) {
      const float* lrow = &lw[(size_t)mrow * lws + (size_t)ncol * R_SZ];
      const float* trow = &tvec[mrow * R_SZ];
      float d = 0.f;
      #pragma unroll
      for (int r2 = 0; r2 < R_SZ; ++r2) d += lrow[r2] * trow[r2];
      v += d;
    }
    if (EPI == EPI_BIAS_GELU || EPI == EPI_LORA_GELU) v = gelu_exact(v);
    if (EPI == EPI_LORA_RESID) v += resid[(size_t)mrow * ldc + ncol];
    C[(size_t)mrow * ldc + ncol] = v;
  }
}

// ---------------------------------------------------------------------------
// LayerNorm over 1024-wide rows; one block (256 threads) per row.
// ---------------------------------------------------------------------------
__global__ void __launch_bounds__(256) ln_kernel(
    const float* __restrict__ e, const float* __restrict__ g,
    const float* __restrict__ b, float* __restrict__ c)
{
  __shared__ float s1[256], s2[256];
  const int row = blockIdx.x, tid = threadIdx.x;
  float v[4], lsum = 0.f, lsq = 0.f;
  #pragma unroll
  for (int i = 0; i < 4; ++i) {
    v[i] = e[(size_t)row * ADA_SZ + tid + i * 256];
    lsum += v[i]; lsq += v[i] * v[i];
  }
  s1[tid] = lsum; s2[tid] = lsq;
  __syncthreads();
  for (int off = 128; off; off >>= 1) {
    if (tid < off) { s1[tid] += s1[tid + off]; s2[tid] += s2[tid + off]; }
    __syncthreads();
  }
  const float mu   = s1[0] * (1.f / ADA_SZ);
  const float var  = s2[0] * (1.f / ADA_SZ) - mu * mu;
  const float rstd = rsqrtf(var + 1e-5f);
  #pragma unroll
  for (int i = 0; i < 4; ++i) {
    const int j = tid + i * 256;
    c[(size_t)row * ADA_SZ + j] = (v[i] - mu) * rstd * g[j] + b[j];
  }
}

// ---------------------------------------------------------------------------
// t[b,r] = sum_d w[b, off + d*8 + r] * v[b,d]   (rank-8 contraction)
// One block (256 threads) per batch row.
// ---------------------------------------------------------------------------
__global__ void __launch_bounds__(256) lora_t_kernel(
    const float* __restrict__ w, int chunk_off,
    const float* __restrict__ v, float* __restrict__ t)
{
  __shared__ float red[256 * R_SZ];
  const int b = blockIdx.x, tid = threadIdx.x;
  float acc[R_SZ] = {};
  const float* wb = w + (size_t)b * WCOLS + chunk_off;
  for (int d = tid; d < D_SZ; d += 256) {
    const float xv = v[(size_t)b * D_SZ + d];
    #pragma unroll
    for (int r = 0; r < R_SZ; ++r) acc[r] += wb[(size_t)d * R_SZ + r] * xv;
  }
  #pragma unroll
  for (int r = 0; r < R_SZ; ++r) red[tid * R_SZ + r] = acc[r];
  __syncthreads();
  for (int off = 128; off; off >>= 1) {
    if (tid < off)
      #pragma unroll
      for (int r = 0; r < R_SZ; ++r)
        red[tid * R_SZ + r] += red[(tid + off) * R_SZ + r];
    __syncthreads();
  }
  if (tid < R_SZ) t[(size_t)b * R_SZ + tid] = red[tid];
}

// ---------------------------------------------------------------------------
extern "C" void kernel_launch(void* const* d_in, const int* in_sizes, int n_in,
                              void* d_out, int out_size, void* d_ws, size_t ws_size,
                              hipStream_t stream)
{
  const float* x         = (const float*)d_in[0];
  const float* ada_emb   = (const float*)d_in[1];
  const float* base_up   = (const float*)d_in[2];
  const float* base_down = (const float*)d_in[3];
  const float* ln_g      = (const float*)d_in[4];
  const float* ln_b      = (const float*)d_in[5];
  const float* W1        = (const float*)d_in[6];
  const float* b1        = (const float*)d_in[7];
  const float* W2        = (const float*)d_in[8];
  const float* b2        = (const float*)d_in[9];
  float* out = (float*)d_out;

  // workspace layout (fp32): c | h | w | y | t2 | t1
  float* c  = (float*)d_ws;
  float* h  = c  + (size_t)B_SZ * ADA_SZ;
  float* w  = h  + (size_t)B_SZ * D_SZ;
  float* y  = w  + (size_t)B_SZ * WCOLS;
  float* t2 = y  + (size_t)B_SZ * D_SZ;
  float* t1 = t2 + (size_t)B_SZ * R_SZ;

  const dim3 blk(256);

  // 1) c = LayerNorm(ada_emb)
  ln_kernel<<<dim3(B_SZ), blk, 0, stream>>>(ada_emb, ln_g, ln_b, c);

  // 2) h = gelu(c @ W1 + b1)
  gemm_bf16_wmma<EPI_BIAS_GELU, false>
      <<<dim3(ADA_SZ / 16), blk, 0, stream>>>(
      c, ADA_SZ, W1, ADA_SZ, b1, h, ADA_SZ, ADA_SZ,
      nullptr, 0, nullptr, nullptr);

  // 3) w = h @ W2 + b2   (the 128 MB streaming GEMM; W2 read exactly once)
  gemm_bf16_wmma<EPI_BIAS, false>
      <<<dim3(WCOLS / 16), blk, 0, stream>>>(
      h, ADA_SZ, W2, WCOLS, b2, w, WCOLS, ADA_SZ,
      nullptr, 0, nullptr, nullptr);

  // 4) t2[b,r] = sum_d la2[b,d,r] * x[b,d]   (la2 = chunk 2 of w)
  lora_t_kernel<<<dim3(B_SZ), blk, 0, stream>>>(w, 2 * D_SZ * R_SZ, x, t2);

  // 5) y = gelu(x @ base_down + lb2 . t2)    (lb2 = chunk 3 of w)
  gemm_bf16_wmma<EPI_LORA_GELU, false>
      <<<dim3(D_SZ / 16), blk, 0, stream>>>(
      x, D_SZ, base_down, D_SZ, nullptr, y, D_SZ, D_SZ,
      w + 3 * D_SZ * R_SZ, WCOLS, t2, nullptr);

  // 6) t1[b,r] = sum_l lb1[b,l,r] * y[b,l]   (lb1 = chunk 1 of w)
  lora_t_kernel<<<dim3(B_SZ), blk, 0, stream>>>(w, 1 * D_SZ * R_SZ, y, t1);

  // 7) out = x + y @ base_up^T + la1 . t1    (la1 = chunk 0 of w)
  gemm_bf16_wmma<EPI_LORA_RESID, true>
      <<<dim3(D_SZ / 16), blk, 0, stream>>>(
      y, D_SZ, base_up, D_SZ, nullptr, out, D_SZ, D_SZ,
      w, WCOLS, t1, x);
}
